// FullKThinningRecurrentNet_81673098101040
// MI455X (gfx1250) — compile-verified, matrix operand order checked by reading, not run on hardware
//
#include <hip/hip_runtime.h>
#include <hip/hip_bf16.h>

typedef _Float16 v16h __attribute__((ext_vector_type(16)));
typedef float    v8f  __attribute__((ext_vector_type(8)));

#define BATCH 2048
#define SEQ   1024
#define NCLS  256
#define HH    128

union BU { uint4 u[2]; v16h h; };
union WU { uint4 u;    _Float16 h[8]; };
union HU { _Float16 h[2]; unsigned int u; };

// CDNA5 hardware tanh (v_tanh_f32) when the builtin exists; else fast exp form.
static __device__ __forceinline__ float fast_tanh(float z) {
#if __has_builtin(__builtin_amdgcn_tanhf)
  return __builtin_amdgcn_tanhf(z);
#elif __has_builtin(__builtin_amdgcn_tanh_f32)
  return __builtin_amdgcn_tanh_f32(z);
#else
  return 1.0f - 2.0f / (__expf(2.0f * z) + 1.0f);
#endif
}

// ---------------------------------------------------------------------------
// Kernel 0: W_ih (f32, [H][I] row-major) -> transposed f16 [I][H] in workspace
// ---------------------------------------------------------------------------
__global__ __launch_bounds__(256) void prep_wih(const float* __restrict__ W_ih,
                                                _Float16* __restrict__ out) {
  int idx = blockIdx.x * 256 + threadIdx.x;      // 0..32767
  int i = idx >> 7;                              // class 0..255
  int h = idx & 127;                             // hidden 0..127
  out[idx] = (_Float16)W_ih[h * NCLS + i];
}

// ---------------------------------------------------------------------------
// Kernel 1: per-row counting sort of x (values in [0,256)) -> u8 sorted rows
// ---------------------------------------------------------------------------
__global__ __launch_bounds__(256) void sort_rows(const int* __restrict__ x,
                                                 unsigned char* __restrict__ xs) {
  __shared__ unsigned int hist[256];
  const int row = blockIdx.x;
  const unsigned t = threadIdx.x;
  hist[t] = 0;
  __syncthreads();
  #pragma unroll
  for (int it = 0; it < 4; ++it) {
    int v = x[row * SEQ + it * 256 + t] & 255;
    atomicAdd(&hist[v], 1u);
  }
  __syncthreads();
  unsigned cnt = hist[t];
  // inclusive Hillis-Steele scan over 256 bins
  for (int off = 1; off < 256; off <<= 1) {
    unsigned val = (t >= (unsigned)off) ? hist[t - off] : 0u;
    __syncthreads();
    hist[t] += val;
    __syncthreads();
  }
  unsigned end = hist[t];
  unsigned start = end - cnt;
  for (unsigned p = start; p < end; ++p)
    xs[row * SEQ + p] = (unsigned char)t;
}

// ---------------------------------------------------------------------------
// Kernel 2: RNN. 1 block = 16 batch rows, 8 waves; wave m owns hidden rows
// 16m..16m+15.  h_new^T = tanh(W_hh @ h^T + Wx^T + bias) via 4x WMMA f16.
// ---------------------------------------------------------------------------
__global__ __launch_bounds__(256) void rnn_kernel(const float* __restrict__ W_hh,
                                                  const float* __restrict__ b_ih,
                                                  const float* __restrict__ b_hh,
                                                  const unsigned char* __restrict__ xs,
                                                  const _Float16* __restrict__ Wih16T,
                                                  float* __restrict__ out) {
  __shared__ __align__(16) unsigned char xs_tile[16 * SEQ];   // 16KB
  __shared__ __align__(16) unsigned int  hbuf[2][1024];       // 2 x 4KB (f16 h^T, B-operand layout)

  const int tid  = threadIdx.x;
  const int wave = tid >> 5;           // 0..7  (= hidden tile m)
  const int lane = tid & 31;
  const int n    = lane & 15;          // batch column within tile
  const int hiHalf = lane >> 4;        // C-layout: lanes 16-31 hold M+8
  const int batch_base = blockIdx.x * 16;
  const int hibase = wave * 16 + hiHalf * 8;   // 8 consecutive hidden rows per lane

  // ---- load W_hh as f16 A-operands (constant across all 1024 steps) ----
  // A 16x32 f16 layout: lanes 0-15: M=lane, K={0..7,16..23}; lanes 16-31: M=lane-16, K={8..15,24..31}
  const int M   = lane & 15;
  const int Klo = hiHalf * 8;
  v16h av[4];
  {
    const float* wrow = W_hh + (wave * 16 + M) * HH;
    #pragma unroll
    for (int k = 0; k < 4; ++k) {
      const float* p = wrow + k * 32 + Klo;
      float4 f0 = ((const float4*)(p     ))[0];
      float4 f1 = ((const float4*)(p +  4))[0];
      float4 f2 = ((const float4*)(p + 16))[0];
      float4 f3 = ((const float4*)(p + 20))[0];
      v16h a;
      a[0]=(_Float16)f0.x; a[1]=(_Float16)f0.y; a[2]=(_Float16)f0.z; a[3]=(_Float16)f0.w;
      a[4]=(_Float16)f1.x; a[5]=(_Float16)f1.y; a[6]=(_Float16)f1.z; a[7]=(_Float16)f1.w;
      a[8]=(_Float16)f2.x; a[9]=(_Float16)f2.y; a[10]=(_Float16)f2.z; a[11]=(_Float16)f2.w;
      a[12]=(_Float16)f3.x; a[13]=(_Float16)f3.y; a[14]=(_Float16)f3.z; a[15]=(_Float16)f3.w;
      av[k] = a;
    }
  }

  // ---- fused bias (b_ih + b_hh) for this lane's 8 hidden rows ----
  float bias[8];
  {
    float4 i0 = ((const float4*)(b_ih + hibase))[0];
    float4 i1 = ((const float4*)(b_ih + hibase + 4))[0];
    float4 h0 = ((const float4*)(b_hh + hibase))[0];
    float4 h1 = ((const float4*)(b_hh + hibase + 4))[0];
    bias[0]=i0.x+h0.x; bias[1]=i0.y+h0.y; bias[2]=i0.z+h0.z; bias[3]=i0.w+h0.w;
    bias[4]=i1.x+h1.x; bias[5]=i1.y+h1.y; bias[6]=i1.z+h1.z; bias[7]=i1.w+h1.w;
  }

  // ---- precompute LDS store slots for h_new (B-operand layout, pairs) ----
  // B 32x16 f16 chunk layout (mirror of A): lanes 0-15: N=lane, K'={0..7,16..23};
  // lanes 16-31: N=lane-16, K'={8..15,24..31}; VGPR j holds K' pair (2 f16 / dword).
  int sidx[4];
  #pragma unroll
  for (int jj = 0; jj < 4; ++jj) {
    int hi  = hibase + 2 * jj;         // even
    int k   = hi >> 5;                 // chunk
    int Kp  = hi & 31;
    int r   = Kp & 15;
    int hi4 = Kp >> 4;
    int laneB, jB;
    if (r < 8) { laneB = n;      jB = (r >> 1) + hi4 * 4; }
    else       { laneB = n + 16; jB = ((r - 8) >> 1) + hi4 * 4; }
    sidx[jj] = k * 256 + laneB * 8 + jB;   // dword index within buffer
  }

  // ---- preload this block's 16 sorted rows (u8) into LDS ----
  {
    const uint4* src = (const uint4*)(xs + (size_t)batch_base * SEQ);
    uint4* dst = (uint4*)xs_tile;
    for (int i = tid; i < 1024; i += 256) dst[i] = src[i];
  }
  // ---- h0 = 0 ----
  for (int i = tid; i < 2048; i += 256) ((unsigned int*)hbuf)[i] = 0u;
  __syncthreads();

  float hv[8];
  int bufc = 0;
  // double-buffered sorted-input word: 4 timesteps per ds_load_b32
  unsigned xw = *(const unsigned int*)(xs_tile + n * SEQ);
  for (int t = 0; t < SEQ; t += 4) {
    // --- issue ALL four Wx gathers up front (addresses depend only on xw);
    //     their L2 latency overlaps the 16 WMMAs + barriers below ---
    WU wu[4];
    #pragma unroll
    for (int s = 0; s < 4; ++s) {
      const unsigned xv = (xw >> (8 * s)) & 255u;
      wu[s].u = ((const uint4*)Wih16T)[xv * 16 + wave * 2 + hiHalf];
    }
    // --- prefetch next macro-iteration's input word (branchless wrap) ---
    const unsigned xw_next =
        *(const unsigned int*)(xs_tile + n * SEQ + ((t + 4) & (SEQ - 1)));

    #pragma unroll
    for (int s = 0; s < 4; ++s) {
      // --- D = W_hh @ h^T : two independent WMMA chains over K=128 ---
      const uint4* bp = (const uint4*)hbuf[bufc];
      v8f acc0 = {}, acc1 = {};
      {
        BU bu;
        bu.u[0] = bp[0 * 64 + lane * 2];
        bu.u[1] = bp[0 * 64 + lane * 2 + 1];
        acc0 = __builtin_amdgcn_wmma_f32_16x16x32_f16(
            false, av[0], false, bu.h, (short)0, acc0, false, false);
      }
      {
        BU bu;
        bu.u[0] = bp[2 * 64 + lane * 2];
        bu.u[1] = bp[2 * 64 + lane * 2 + 1];
        acc1 = __builtin_amdgcn_wmma_f32_16x16x32_f16(
            false, av[2], false, bu.h, (short)0, acc1, false, false);
      }
      {
        BU bu;
        bu.u[0] = bp[1 * 64 + lane * 2];
        bu.u[1] = bp[1 * 64 + lane * 2 + 1];
        acc0 = __builtin_amdgcn_wmma_f32_16x16x32_f16(
            false, av[1], false, bu.h, (short)0, acc0, false, false);
      }
      {
        BU bu;
        bu.u[0] = bp[3 * 64 + lane * 2];
        bu.u[1] = bp[3 * 64 + lane * 2 + 1];
        acc1 = __builtin_amdgcn_wmma_f32_16x16x32_f16(
            false, av[3], false, bu.h, (short)0, acc1, false, false);
      }

      // --- h = tanh(acc + Wx + bias) via CDNA5 v_tanh_f32 ---
      #pragma unroll
      for (int j = 0; j < 8; ++j) {
        float z = (acc0[j] + acc1[j]) + ((float)wu[s].h[j] + bias[j]);
        hv[j] = fast_tanh(z);
      }

      // --- store h_new^T (f16 pairs) into the other buffer ---
      const int bufn = bufc ^ 1;
      #pragma unroll
      for (int jj = 0; jj < 4; ++jj) {
        HU p;
        p.h[0] = (_Float16)hv[2 * jj];
        p.h[1] = (_Float16)hv[2 * jj + 1];
        hbuf[bufn][sidx[jj]] = p.u;
      }
      __syncthreads();
      bufc = bufn;
    }
    xw = xw_next;
  }

  // ---- final h: 8 consecutive floats per lane ----
  float* orow = out + (size_t)(batch_base + n) * HH + hibase;
  ((float4*)orow)[0]       = make_float4(hv[0], hv[1], hv[2], hv[3]);
  ((float4*)(orow + 4))[0] = make_float4(hv[4], hv[5], hv[6], hv[7]);
}

// ---------------------------------------------------------------------------
extern "C" void kernel_launch(void* const* d_in, const int* in_sizes, int n_in,
                              void* d_out, int out_size, void* d_ws, size_t ws_size,
                              hipStream_t stream) {
  const int*   x    = (const int*)d_in[0];
  const float* W_ih = (const float*)d_in[1];
  const float* W_hh = (const float*)d_in[2];
  const float* b_ih = (const float*)d_in[3];
  const float* b_hh = (const float*)d_in[4];
  float* out = (float*)d_out;

  unsigned char* xs     = (unsigned char*)d_ws;                       // 2 MB
  _Float16*      Wih16T = (_Float16*)((char*)d_ws + (2u << 20));      // 64 KB

  prep_wih <<<128, 256, 0, stream>>>(W_ih, Wih16T);
  sort_rows<<<BATCH, 256, 0, stream>>>(x, xs);
  rnn_kernel<<<BATCH / 16, 256, 0, stream>>>(W_hh, b_ih, b_hh, xs, Wih16T, out);
}
